// SelfAttention_22754736734625
// MI455X (gfx1250) — matrix operand order repaired and checked by reading, not compiled, not run on hardware
//
#include <hip/hip_runtime.h>
#include <hip/hip_bf16.h>

// ---------------------------------------------------------------------------
// MI455X (gfx1250) fused self-attention.  B=2, T=4096, K=1024, H=16, d=64.
//  * one-time fp32->bf16 cast of x and weights (removes cvt from hot loops)
//  * tile staging (all but attention V) via GLOBAL_LOAD_ASYNC_TO_LDS_B128 (ASYNCcnt), double
//    buffered: 1 barrier + 1 s_wait_asynccnt per K-tile
//  * all math via v_wmma_f32_16x16x32_bf16 (fp32 accumulate)
//  * flash-attention fused softmax (scores never hit HBM)
// Workspace (bf16): q|k|v|y (4x16MB) + x_bf (16MB) + Wq,Wk,Wv,Wu_bf (4x2MB)
//                 = 88 MB.
// ---------------------------------------------------------------------------

typedef __bf16 bf16;
typedef __attribute__((ext_vector_type(8)))  __bf16 v8bf;
typedef __attribute__((ext_vector_type(16))) __bf16 v16bf;
typedef __attribute__((ext_vector_type(8)))  float  v8f;

#define KDIM 1024
#define TDIM 4096
#define HDIM 16
#define DDIM 64
#define MTOT 8192   // B*T

__device__ __forceinline__ v8f wmma_bf16(v16bf a, v16bf b, v8f c) {
  return __builtin_amdgcn_wmma_f32_16x16x32_bf16(false, a, false, b,
                                                 (short)0, c, false, false);
}

// Async DMA: global -> LDS, 16 bytes per lane, tracked by ASYNCcnt.
__device__ __forceinline__ void async_copy_b128(const void* g, void* l) {
  unsigned lds = (unsigned)(unsigned long long)l;          // LDS byte offset
  unsigned long long ga = (unsigned long long)g;
  asm volatile("global_load_async_to_lds_b128 %0, %1, off"
               :: "v"(lds), "v"(ga) : "memory");
}
__device__ __forceinline__ void wait_async0() {
  asm volatile("s_wait_asynccnt 0" ::: "memory");
}

// A-fragment (16x32 bf16): lane m=l%16 holds row m.
// lanes 0-15: K={0..7,16..23}; lanes 16-31: K={8..15,24..31}
__device__ __forceinline__ v16bf load_frag_a(const bf16* row_ptr, int lane) {
  const int base = (lane < 16) ? 0 : 8;
  v8bf lo = *(const v8bf*)(row_ptr + base);
  v8bf hi = *(const v8bf*)(row_ptr + 16 + base);
  v16bf a;
#pragma unroll
  for (int i = 0; i < 8; ++i) { a[i] = lo[i]; a[i + 8] = hi[i]; }
  return a;
}

// B-fragment (32x16 bf16): lane n=l%16; lanes 0-15 K={0..7,16..23}, 16-31 K={8..15,24..31}.
__device__ __forceinline__ v16bf load_frag_b(const bf16* col_ptr, int lane) {
  const int base = (lane < 16) ? 0 : 8;
  v8bf lo = *(const v8bf*)(col_ptr + base);
  v8bf hi = *(const v8bf*)(col_ptr + 16 + base);
  v16bf b;
#pragma unroll
  for (int i = 0; i < 8; ++i) { b[i] = lo[i]; b[i + 8] = hi[i]; }
  return b;
}

// ---------------------------------------------------------------------------
// Kernel 0: one-time fp32 -> bf16 cast (one float4 per thread).
// ---------------------------------------------------------------------------
__global__ __launch_bounds__(256) void cast_f32_bf16_kernel(
    const float* __restrict__ src, bf16* __restrict__ dst, int n4) {
  const int i = blockIdx.x * 256 + threadIdx.x;
  if (i < n4) {
    float4 d = ((const float4*)src)[i];
    bf16* o = dst + (size_t)i * 4;
    o[0] = (bf16)d.x; o[1] = (bf16)d.y; o[2] = (bf16)d.z; o[3] = (bf16)d.w;
  }
}

// ---------------------------------------------------------------------------
// Kernel 1: QKV projection.  out = x @ W^T + b, bf16, layout [b,h,t,d].
// Tile 128(M) x 64(N) x 64(K-step); 8 waves = 4(M) x 2(N), 32x32 per wave.
// Double-buffered async LDS staging.
// ---------------------------------------------------------------------------
#define G_BM 128
#define G_BN 64
#define G_BK 64
#define G_LDK 72   // 64 + 8 pad (144 B row stride, 16B multiple)

__global__ __launch_bounds__(256) void qkv_gemm_kernel(
    const bf16* __restrict__ xb,
    const bf16* __restrict__ wqb, const float* __restrict__ bq,
    const bf16* __restrict__ wkb, const float* __restrict__ bk,
    const bf16* __restrict__ wvb, const float* __restrict__ bv,
    bf16* __restrict__ qo, bf16* __restrict__ ko, bf16* __restrict__ vo) {
  __shared__ __align__(16) bf16 xs[2 * G_BM * G_LDK];
  __shared__ __align__(16) bf16 ws2[2 * G_BN * G_LDK];

  const int tid = threadIdx.x;
  const int lane = tid & 31;
  const int wave = tid >> 5;
  const int wm = wave & 3;
  const int wn = wave >> 2;
  const int m0 = blockIdx.x * G_BM;
  const int n0 = blockIdx.y * G_BN;

  const bf16* W;
  const float* bias;
  bf16* out;
  if (blockIdx.z == 0)      { W = wqb; bias = bq; out = qo; }
  else if (blockIdx.z == 1) { W = wkb; bias = bk; out = ko; }
  else                      { W = wvb; bias = bv; out = vo; }

  auto stage = [&](int kt, int buf) {
    bf16* xd = xs + buf * (G_BM * G_LDK);
#pragma unroll
    for (int i = 0; i < 4; ++i) {           // 128x64 bf16 = 1024 16B chunks
      int f = i * 256 + tid;
      int r = f >> 3, c = (f & 7) << 3;
      async_copy_b128(xb + (size_t)(m0 + r) * KDIM + kt + c,
                      xd + r * G_LDK + c);
    }
    bf16* wd = ws2 + buf * (G_BN * G_LDK);
#pragma unroll
    for (int i = 0; i < 2; ++i) {           // 64x64 bf16 = 512 16B chunks
      int f = i * 256 + tid;
      int r = f >> 3, c = (f & 7) << 3;
      async_copy_b128(W + (size_t)(n0 + r) * KDIM + kt + c,
                      wd + r * G_LDK + c);
    }
  };

  v8f acc[2][2] = {};
  const int arow = wm * 32 + (lane & 15);
  const int brow = wn * 32 + (lane & 15);

  stage(0, 0);
  for (int kt = 0; kt < KDIM; kt += G_BK) {
    const int buf = (kt >> 6) & 1;
    wait_async0();        // my async stores to LDS for tile kt are done
    __syncthreads();      // everyone's are; previous tile's readers done too
    if (kt + G_BK < KDIM) stage(kt + G_BK, buf ^ 1);

    const bf16* xt = xs + buf * (G_BM * G_LDK);
    const bf16* wt = ws2 + buf * (G_BN * G_LDK);
#pragma unroll
    for (int ks = 0; ks < G_BK; ks += 32) {
      v16bf a0 = load_frag_a(xt + arow * G_LDK + ks, lane);
      v16bf a1 = load_frag_a(xt + (arow + 16) * G_LDK + ks, lane);
      v16bf b0 = load_frag_b(wt + brow * G_LDK + ks, lane);
      v16bf b1 = load_frag_b(wt + (brow + 16) * G_LDK + ks, lane);
      acc[0][0] = wmma_bf16(a0, b0, acc[0][0]);
      acc[0][1] = wmma_bf16(a0, b1, acc[0][1]);
      acc[1][0] = wmma_bf16(a1, b0, acc[1][0]);
      acc[1][1] = wmma_bf16(a1, b1, acc[1][1]);
    }
  }

  // Epilogue: +bias, cast bf16, scatter into [b,h,t,d].
  const int hi = lane >> 4;
#pragma unroll
  for (int i = 0; i < 2; ++i) {
#pragma unroll
    for (int j = 0; j < 2; ++j) {
      const int ncol = n0 + wn * 32 + j * 16 + (lane & 15);
      const int h = ncol >> 6, dc = ncol & 63;
      const float bb = bias[ncol];
#pragma unroll
      for (int r = 0; r < 8; ++r) {
        const int mrow = m0 + wm * 32 + i * 16 + r + hi * 8;
        const int bI = mrow >> 12;
        const int t  = mrow & (TDIM - 1);
        out[(((size_t)(bI * HDIM + h)) * TDIM + t) * DDIM + dc] =
            (bf16)(acc[i][j][r] + bb);
      }
    }
  }
}

// ---------------------------------------------------------------------------
// Kernel 2: flash attention.  Block = 128 queries of one (b,h); 8 waves x 16
// query rows; key blocks of 64, double-buffered (async K, transposed V).
// ---------------------------------------------------------------------------
#define A_KB 64
#define A_LDS 72

__device__ __forceinline__ float half_reduce_max(float v) {
#pragma unroll
  for (int off = 8; off >= 1; off >>= 1)
    v = fmaxf(v, __shfl_xor(v, off, 32));
  return v;
}
__device__ __forceinline__ float half_reduce_sum(float v) {
#pragma unroll
  for (int off = 8; off >= 1; off >>= 1)
    v += __shfl_xor(v, off, 32);
  return v;
}

__global__ __launch_bounds__(256) void attn_kernel(
    const bf16* __restrict__ q, const bf16* __restrict__ k,
    const bf16* __restrict__ v, bf16* __restrict__ y) {
  __shared__ __align__(16) bf16 ks_s[2 * A_KB * A_LDS];   // [key][d]
  __shared__ __align__(16) bf16 vt_s[2 * DDIM * A_LDS];   // [d][key]
  __shared__ __align__(16) bf16 ps[8 * 16 * A_LDS];       // per-wave P

  const int tid = threadIdx.x;
  const int lane = tid & 31;
  const int wave = tid >> 5;
  const int bh = blockIdx.y;
  const size_t head_off = (size_t)bh * TDIM * DDIM;
  const bf16* qb = q + head_off;
  const bf16* kb = k + head_off;
  const bf16* vb = v + head_off;

  // Resident Q fragments for this wave's 16 query rows.
  const int qrow = blockIdx.x * 128 + wave * 16 + (lane & 15);
  v16bf qa0 = load_frag_a(qb + (size_t)qrow * DDIM + 0, lane);
  v16bf qa1 = load_frag_a(qb + (size_t)qrow * DDIM + 32, lane);

  v8f o[4] = {};
  float mstat[8], lstat[8];
#pragma unroll
  for (int r = 0; r < 8; ++r) { mstat[r] = -INFINITY; lstat[r] = 0.0f; }
  const float scale = 0.03125f;   // 1/sqrt(1024)
  bf16* myp = ps + wave * 16 * A_LDS;
  const int hi = lane >> 4;

  auto stage = [&](int kb0, int buf) {
    bf16* kd = ks_s + buf * (A_KB * A_LDS);
#pragma unroll
    for (int i = 0; i < 2; ++i) {           // async K: 512 16B chunks
      int f = i * 256 + tid;
      int r = f >> 3, c = (f & 7) << 3;
      async_copy_b128(kb + (size_t)(kb0 + r) * DDIM + c, kd + r * A_LDS + c);
    }
    bf16* vd = vt_s + buf * (DDIM * A_LDS);
#pragma unroll
    for (int i = 0; i < 2; ++i) {           // V transpose: VGPR staged
      int f = i * 256 + tid;
      int r = f >> 3, c = (f & 7) << 3;
      v8bf d = *(const v8bf*)(vb + (size_t)(kb0 + r) * DDIM + c);
#pragma unroll
      for (int e = 0; e < 8; ++e) vd[(c + e) * A_LDS + r] = d[e];
    }
  };

  stage(0, 0);
  for (int it = 0; it < TDIM / A_KB; ++it) {
    const int buf = it & 1;
    wait_async0();
    __syncthreads();
    if (it + 1 < TDIM / A_KB) stage((it + 1) * A_KB, buf ^ 1);

    const bf16* kt = ks_s + buf * (A_KB * A_LDS);
    const bf16* vt = vt_s + buf * (DDIM * A_LDS);

    // S = Q @ K^T : 16 x 64 per wave.
    v8f s[4] = {};
#pragma unroll
    for (int g = 0; g < 4; ++g) {
      const int krow = g * 16 + (lane & 15);
      v16bf bk0 = load_frag_b(kt + krow * A_LDS + 0, lane);
      v16bf bk1 = load_frag_b(kt + krow * A_LDS + 32, lane);
      s[g] = wmma_bf16(qa0, bk0, s[g]);
      s[g] = wmma_bf16(qa1, bk1, s[g]);
    }

    // Online softmax (row = r + 8*hi, striped over a half-wave).
#pragma unroll
    for (int r = 0; r < 8; ++r) {
      float rmax = fmaxf(fmaxf(s[0][r], s[1][r]), fmaxf(s[2][r], s[3][r]));
      rmax = half_reduce_max(rmax) * scale;
      const float mnew = fmaxf(mstat[r], rmax);
      const float alpha = __expf(mstat[r] - mnew);
      float pv0 = __expf(s[0][r] * scale - mnew);
      float pv1 = __expf(s[1][r] * scale - mnew);
      float pv2 = __expf(s[2][r] * scale - mnew);
      float pv3 = __expf(s[3][r] * scale - mnew);
      const float rsum = half_reduce_sum(pv0 + pv1 + pv2 + pv3);
      mstat[r] = mnew;
      lstat[r] = lstat[r] * alpha + rsum;
#pragma unroll
      for (int j = 0; j < 4; ++j) o[j][r] *= alpha;
      // Spill P row to wave-private LDS (in-wave DS ordering is in-order).
      const int prow = r + hi * 8;
      bf16* pr = myp + prow * A_LDS + (lane & 15);
      pr[0]  = (bf16)pv0;
      pr[16] = (bf16)pv1;
      pr[32] = (bf16)pv2;
      pr[48] = (bf16)pv3;
    }

    // O += P @ V.
    v16bf pa0 = load_frag_a(myp + (lane & 15) * A_LDS + 0, lane);
    v16bf pa1 = load_frag_a(myp + (lane & 15) * A_LDS + 32, lane);
#pragma unroll
    for (int j = 0; j < 4; ++j) {
      const int drow = j * 16 + (lane & 15);
      v16bf bv0 = load_frag_b(vt + drow * A_LDS + 0, lane);
      v16bf bv1 = load_frag_b(vt + drow * A_LDS + 32, lane);
      o[j] = wmma_bf16(pa0, bv0, o[j]);
      o[j] = wmma_bf16(pa1, bv1, o[j]);
    }
  }

  // Normalize, write y bf16 in [b*T+t][h*64+d] (GEMM-ready layout).
  const int bI = bh >> 4, h = bh & 15;
#pragma unroll
  for (int j = 0; j < 4; ++j) {
    const int dc = j * 16 + (lane & 15);
#pragma unroll
    for (int r = 0; r < 8; ++r) {
      const int t = blockIdx.x * 128 + wave * 16 + r + hi * 8;
      const float val = o[j][r] / lstat[r];
      y[((size_t)(bI * TDIM + t)) * KDIM + h * DDIM + dc] = (bf16)val;
    }
  }
}

// ---------------------------------------------------------------------------
// Kernel 3: output projection.  out = y @ Wu^T + bu (fp32 out).
// ---------------------------------------------------------------------------
__global__ __launch_bounds__(256) void out_gemm_kernel(
    const bf16* __restrict__ yb, const bf16* __restrict__ wub,
    const float* __restrict__ bu, float* __restrict__ out) {
  __shared__ __align__(16) bf16 xs[2 * G_BM * G_LDK];
  __shared__ __align__(16) bf16 ws2[2 * G_BN * G_LDK];

  const int tid = threadIdx.x;
  const int lane = tid & 31;
  const int wave = tid >> 5;
  const int wm = wave & 3;
  const int wn = wave >> 2;
  const int m0 = blockIdx.x * G_BM;
  const int n0 = blockIdx.y * G_BN;

  auto stage = [&](int kt, int buf) {
    bf16* xd = xs + buf * (G_BM * G_LDK);
#pragma unroll
    for (int i = 0; i < 4; ++i) {
      int f = i * 256 + tid;
      int r = f >> 3, c = (f & 7) << 3;
      async_copy_b128(yb + (size_t)(m0 + r) * KDIM + kt + c,
                      xd + r * G_LDK + c);
    }
    bf16* wd = ws2 + buf * (G_BN * G_LDK);
#pragma unroll
    for (int i = 0; i < 2; ++i) {
      int f = i * 256 + tid;
      int r = f >> 3, c = (f & 7) << 3;
      async_copy_b128(wub + (size_t)(n0 + r) * KDIM + kt + c,
                      wd + r * G_LDK + c);
    }
  };

  v8f acc[2][2] = {};
  const int arow = wm * 32 + (lane & 15);
  const int brow = wn * 32 + (lane & 15);

  stage(0, 0);
  for (int kt = 0; kt < KDIM; kt += G_BK) {
    const int buf = (kt >> 6) & 1;
    wait_async0();
    __syncthreads();
    if (kt + G_BK < KDIM) stage(kt + G_BK, buf ^ 1);

    const bf16* xt = xs + buf * (G_BM * G_LDK);
    const bf16* wt = ws2 + buf * (G_BN * G_LDK);
#pragma unroll
    for (int ks = 0; ks < G_BK; ks += 32) {
      v16bf a0 = load_frag_a(xt + arow * G_LDK + ks, lane);
      v16bf a1 = load_frag_a(xt + (arow + 16) * G_LDK + ks, lane);
      v16bf b0 = load_frag_b(wt + brow * G_LDK + ks, lane);
      v16bf b1 = load_frag_b(wt + (brow + 16) * G_LDK + ks, lane);
      acc[0][0] = wmma_bf16(a0, b0, acc[0][0]);
      acc[0][1] = wmma_bf16(a0, b1, acc[0][1]);
      acc[1][0] = wmma_bf16(a1, b0, acc[1][0]);
      acc[1][1] = wmma_bf16(a1, b1, acc[1][1]);
    }
  }

  const int hi = lane >> 4;
#pragma unroll
  for (int i = 0; i < 2; ++i) {
#pragma unroll
    for (int j = 0; j < 2; ++j) {
      const int ncol = n0 + wn * 32 + j * 16 + (lane & 15);
      const float bb = bu[ncol];
#pragma unroll
      for (int r = 0; r < 8; ++r) {
        const int mrow = m0 + wm * 32 + i * 16 + r + hi * 8;
        out[(size_t)mrow * KDIM + ncol] = acc[i][j][r] + bb;
      }
    }
  }
}

// ---------------------------------------------------------------------------
extern "C" void kernel_launch(void* const* d_in, const int* in_sizes, int n_in,
                              void* d_out, int out_size, void* d_ws,
                              size_t ws_size, hipStream_t stream) {
  (void)in_sizes; (void)n_in; (void)out_size; (void)ws_size;
  const float* x  = (const float*)d_in[0];
  const float* Wq = (const float*)d_in[1];
  const float* bq = (const float*)d_in[2];
  const float* Wk = (const float*)d_in[3];
  const float* bk = (const float*)d_in[4];
  const float* Wv = (const float*)d_in[5];
  const float* bv = (const float*)d_in[6];
  const float* Wu = (const float*)d_in[7];
  const float* bu = (const float*)d_in[8];
  float* out = (float*)d_out;

  const size_t xe = (size_t)MTOT * KDIM;   // 8 M elements
  const size_t we = (size_t)KDIM * KDIM;   // 1 M elements
  bf16* qws = (bf16*)d_ws;
  bf16* kws = qws + xe;
  bf16* vws = kws + xe;
  bf16* yws = vws + xe;
  bf16* xbb = yws + xe;
  bf16* wqb = xbb + xe;
  bf16* wkb = wqb + we;
  bf16* wvb = wkb + we;
  bf16* wub = wvb + we;                    // total 88 MB

  // 0) One-time bf16 casts.
  cast_f32_bf16_kernel<<<(int)(xe / 4 / 256), 256, 0, stream>>>(x, xbb,
                                                               (int)(xe / 4));
  cast_f32_bf16_kernel<<<(int)(we / 4 / 256), 256, 0, stream>>>(Wq, wqb,
                                                               (int)(we / 4));
  cast_f32_bf16_kernel<<<(int)(we / 4 / 256), 256, 0, stream>>>(Wk, wkb,
                                                               (int)(we / 4));
  cast_f32_bf16_kernel<<<(int)(we / 4 / 256), 256, 0, stream>>>(Wv, wvb,
                                                               (int)(we / 4));
  cast_f32_bf16_kernel<<<(int)(we / 4 / 256), 256, 0, stream>>>(Wu, wub,
                                                               (int)(we / 4));

  // 1) QKV projections (z selects Wq/Wk/Wv).
  qkv_gemm_kernel<<<dim3(MTOT / G_BM, KDIM / G_BN, 3), 256, 0, stream>>>(
      xbb, wqb, bq, wkb, bk, wvb, bv, qws, kws, vws);

  // 2) Flash attention.
  attn_kernel<<<dim3(TDIM / 128, 2 * HDIM), 256, 0, stream>>>(
      qws, kws, vws, yws);

  // 3) Output projection.
  out_gemm_kernel<<<dim3(MTOT / G_BM, KDIM / G_BN), 256, 0, stream>>>(
      yws, Wu != nullptr ? wub : wub, bu, out);
}
